// HyperbolicAttention_49271864819920
// MI455X (gfx1250) — matrix operand order repaired and checked by reading, not compiled, not run on hardware
//
#include <hip/hip_runtime.h>
#include <math.h>

// ---------------------------------------------------------------------------
// Hyperbolic attention, B=1 N=256 D=512 H=8 h=64, fp32 in/out.
// All matmul stages run on v_wmma_f32_16x16x32_bf16 (wave32, 16x16 tile/wave).
// ---------------------------------------------------------------------------

#define EPSF 1e-7f

typedef __attribute__((ext_vector_type(16))) __bf16 v16bf;
typedef __attribute__((ext_vector_type(8)))  float  v8f;

static constexpr int N = 256, D = 512, H = 8, HD = 64, QKVN = 1536;

// ---- WMMA fragment loaders (ISA 7.12.2 layouts) ---------------------------
// A (16x32 bf16, MxK): lane L holds row M=L%16; elems 0..7  -> K = kb+0..7,
// elems 8..15 -> K = kb+16..23, where kb = (L/16)*8.
static __device__ inline v16bf load_a_frag(const __bf16* A, int lda, int row0,
                                           int k0, int lane) {
  const __bf16* p = A + (row0 + (lane & 15)) * lda + k0 + ((lane >> 4) << 3);
  v16bf f;
#pragma unroll
  for (int e = 0; e < 8; ++e) f[e] = p[e];
#pragma unroll
  for (int e = 0; e < 8; ++e) f[8 + e] = p[16 + e];
  return f;
}

// B (32x16 bf16, KxN), stored TRANSPOSED in ws as Bt[N][K]: lane L holds
// col N=L%16; elems 0..7 -> K = kb+0..7, elems 8..15 -> K = kb+16..23,
// where kb = (L/16)*8 (same K order as A).
static __device__ inline v16bf load_bt_frag(const __bf16* Bt, int ldb, int col0,
                                            int k0, int lane) {
  const __bf16* p = Bt + (col0 + (lane & 15)) * ldb + k0 + ((lane >> 4) << 3);
  v16bf f;
#pragma unroll
  for (int e = 0; e < 8; ++e) f[e] = p[e];
#pragma unroll
  for (int e = 0; e < 8; ++e) f[8 + e] = p[16 + e];
  return f;
}

// C/D (16x16 f32): VGPR r, lane L -> row = r + (L/16)*8, col = L%16.
static __device__ inline v8f wmma_tile(const __bf16* A, int lda,
                                       const __bf16* Bt, int ldb,
                                       int row0, int col0, int K, int lane) {
  v8f acc = {0.f, 0.f, 0.f, 0.f, 0.f, 0.f, 0.f, 0.f};
  for (int k = 0; k < K; k += 32) {
    v16bf a = load_a_frag(A, lda, row0, k, lane);
    v16bf b = load_bt_frag(Bt, ldb, col0, k, lane);
    acc = __builtin_amdgcn_wmma_f32_16x16x32_bf16(false, a, false, b,
                                                  (short)0, acc, false, false);
  }
  return acc;
}

// ---- K0: weights -> bf16, transposed (Bt[N][K] layout) --------------------
__global__ void k_prep_weights(const float* __restrict__ Wqkv,
                               const float* __restrict__ Wout,
                               __bf16* __restrict__ WqkvT,
                               __bf16* __restrict__ WoutT) {
  int idx = blockIdx.x * blockDim.x + threadIdx.x;
  if (idx < QKVN * D) {           // WqkvT[j][k] = Wqkv[k][j]
    int j = idx / D, k = idx % D;
    WqkvT[idx] = (__bf16)Wqkv[k * QKVN + j];
  }
  if (idx < D * D) {              // WoutT[j][k] = Wout[k][j]
    int j = idx / D, k = idx % D;
    WoutT[idx] = (__bf16)Wout[k * D + j];
  }
}

// ---- K1: x_t = logmap0(x_hyp, c) -> bf16 ----------------------------------
__global__ void k_logmap0(const float* __restrict__ x,
                          const float* __restrict__ c_ptr,
                          __bf16* __restrict__ x_bf) {
  __shared__ float red[256];
  int n = blockIdx.x, t = threadIdx.x;
  const float* row = x + n * D;
  float a0 = row[t], a1 = row[t + 256];
  red[t] = a0 * a0 + a1 * a1;
  __syncthreads();
  for (int s = 128; s > 0; s >>= 1) {
    if (t < s) red[t] += red[t + s];
    __syncthreads();
  }
  float nrm = sqrtf(red[0]);
  float sc = fmaxf(sqrtf(c_ptr[0]), EPSF);
  float mag = atanhf(fminf(nrm, 1.f - EPSF)) / sc;
  float s = (nrm < EPSF) ? 0.f : mag / fmaxf(nrm, EPSF);
  x_bf[n * D + t] = (__bf16)(a0 * s);
  x_bf[n * D + t + 256] = (__bf16)(a1 * s);
}

// ---- K2: qkv = x_t @ Wqkv + bqkv  (256x1536, K=512) -----------------------
__global__ void k_gemm_qkv(const __bf16* __restrict__ A,
                           const __bf16* __restrict__ Bt,
                           const float* __restrict__ bias,
                           float* __restrict__ C) {
  int wave = blockIdx.x * (blockDim.x >> 5) + (threadIdx.x >> 5);
  int lane = threadIdx.x & 31;
  int tm = wave / (QKVN / 16), tn = wave % (QKVN / 16);
  v8f acc = wmma_tile(A, D, Bt, D, tm * 16, tn * 16, D, lane);
  int col = tn * 16 + (lane & 15);
  float b = bias[col];
  int rbase = tm * 16 + ((lane >> 4) << 3);
#pragma unroll
  for (int r = 0; r < 8; ++r) C[(rbase + r) * QKVN + col] = acc[r] + b;
}

// ---- K3: RoPE + expmap0 per (head, n); also v -> vT bf16 ------------------
__global__ void k_rope_expmap(const float* __restrict__ qkv,
                              const float* __restrict__ freqs,
                              const float* __restrict__ c_logits,
                              __bf16* __restrict__ q_hyp,
                              __bf16* __restrict__ k_hyp,
                              __bf16* __restrict__ vT,
                              float* __restrict__ q2,
                              float* __restrict__ k2) {
  __shared__ float redq[64];
  __shared__ float redk[64];
  int b = blockIdx.x;              // hd*256 + n
  int hd = b >> 8, n = b & 255;
  int d = threadIdx.x;             // 0..63
  const float* base = qkv + n * QKVN + hd * HD;
  float c = log1pf(expf(c_logits[hd]));
  float sc = fmaxf(sqrtf(c), EPSF);
  int dp = d & 31;
  float fr = freqs[n * 32 + dp];
  float cs = cosf(fr), sn = sinf(fr);
  float qr = base[dp], qi = base[32 + dp];
  float kr = base[D + dp], ki = base[D + 32 + dp];
  float qrot = (d < 32) ? qr * cs - qi * sn : qr * sn + qi * cs;
  float krot = (d < 32) ? kr * cs - ki * sn : kr * sn + ki * cs;
  float vv = base[2 * D + d];
  redq[d] = qrot * qrot;
  redk[d] = krot * krot;
  __syncthreads();
  for (int s = 32; s > 0; s >>= 1) {
    if (d < s) { redq[d] += redq[d + s]; redk[d] += redk[d + s]; }
    __syncthreads();
  }
  float qn = sqrtf(redq[0]), kn = sqrtf(redk[0]);
  // expmap0 + project: result norm = mag = tanh(sc*n)/sc, clamped to 1-EPS
  float qsafe = fmaxf(qn, EPSF), ksafe = fmaxf(kn, EPSF);
  float qmag = tanhf(sc * qsafe) / sc, kmag = tanhf(sc * ksafe) / sc;
  float qproj = (fmaxf(qmag, EPSF) >= 1.f) ? (1.f - EPSF) / qmag : 1.f;
  float kproj = (fmaxf(kmag, EPSF) >= 1.f) ? (1.f - EPSF) / kmag : 1.f;
  float qs = (qn < EPSF) ? 0.f : qmag * qproj / qsafe;
  float ks = (kn < EPSF) ? 0.f : kmag * kproj / ksafe;
  q_hyp[(hd * N + n) * HD + d] = (__bf16)(qrot * qs);
  k_hyp[(hd * N + n) * HD + d] = (__bf16)(krot * ks);
  vT[(hd * HD + d) * N + n] = (__bf16)vv;   // [H][64][N] for B-frag loads
  if (d == 0) {
    float qf = (qn < EPSF) ? 0.f : qmag * qproj;
    float kf = (kn < EPSF) ? 0.f : kmag * kproj;
    q2[hd * N + n] = qf * qf;
    k2[hd * N + n] = kf * kf;
  }
}

// ---- K4a: per-head Gram S = q_hyp @ k_hyp^T (256x256, K=64) ---------------
__global__ void k_gram(const __bf16* __restrict__ q_hyp,
                       const __bf16* __restrict__ k_hyp,
                       float* __restrict__ S) {
  int wave = blockIdx.x * (blockDim.x >> 5) + (threadIdx.x >> 5);
  int lane = threadIdx.x & 31;
  int hd = wave >> 8;
  int t = wave & 255, ti = t >> 4, tj = t & 15;
  const __bf16* A = q_hyp + hd * N * HD;
  const __bf16* Bt = k_hyp + hd * N * HD;   // rows j contiguous over d
  v8f acc = wmma_tile(A, HD, Bt, HD, ti * 16, tj * 16, HD, lane);
  float* Sh = S + hd * N * N;
  int col = tj * 16 + (lane & 15);
  int rbase = ti * 16 + ((lane >> 4) << 3);
#pragma unroll
  for (int r = 0; r < 8; ++r) Sh[(rbase + r) * N + col] = acc[r];
}

// ---- K4b: mobius distance + causal mask + softmax -> w (bf16) -------------
__global__ void k_scores_softmax(const float* __restrict__ S,
                                 const float* __restrict__ q2a,
                                 const float* __restrict__ k2a,
                                 const float* __restrict__ c_logits,
                                 const float* __restrict__ geo_scale,
                                 __bf16* __restrict__ W) {
  __shared__ float red[256];
  int b = blockIdx.x;              // hd*256 + i
  int hd = b >> 8, i = b & 255;
  int j = threadIdx.x;
  float c = log1pf(expf(c_logits[hd]));
  float sc = fmaxf(sqrtf(c), EPSF);
  float x2 = q2a[hd * N + i];
  float y2 = k2a[hd * N + j];
  float xy = -S[(hd * N + i) * N + j];      // x = -q_hyp
  float Am = 1.f + 2.f * c * xy + c * y2;
  float Bm = 1.f - c * x2;
  float num2 = Am * Am * x2 + Bm * Bm * y2 + 2.f * Am * Bm * xy;
  float den = fmaxf(1.f + 2.f * c * xy + c * c * x2 * y2, EPSF);
  float nrm = sqrtf(fmaxf(num2, 0.f)) / den;
  float addn = (fmaxf(nrm, EPSF) >= 1.f) ? (1.f - EPSF) : nrm;  // _project
  float tt = fminf(sc * addn, 1.f - EPSF);
  float dist = 2.f * atanhf(tt) / sc;
  float score = (j <= i) ? -geo_scale[hd] * dist : -INFINITY;
  red[j] = score;
  __syncthreads();
  for (int s = 128; s > 0; s >>= 1) {
    if (j < s) red[j] = fmaxf(red[j], red[j + s]);
    __syncthreads();
  }
  float mx = red[0];
  __syncthreads();
  float e = (j <= i) ? expf(score - mx) : 0.f;
  red[j] = e;
  __syncthreads();
  for (int s = 128; s > 0; s >>= 1) {
    if (j < s) red[j] += red[j + s];
    __syncthreads();
  }
  W[(hd * N + i) * N + j] = (__bf16)(e / red[0]);
}

// ---- K5: ctx = w @ v per head (256x64, K=256) -> ctx bf16 [256][512] ------
__global__ void k_pv(const __bf16* __restrict__ W,
                     const __bf16* __restrict__ vT,
                     __bf16* __restrict__ ctx) {
  int wave = blockIdx.x * (blockDim.x >> 5) + (threadIdx.x >> 5);
  int lane = threadIdx.x & 31;
  int hd = wave >> 6;
  int t = wave & 63, ti = t >> 2, td = t & 3;
  const __bf16* A = W + hd * N * N;
  const __bf16* Bt = vT + hd * HD * N;      // vT[d][j]
  v8f acc = wmma_tile(A, N, Bt, N, ti * 16, td * 16, N, lane);
  int col = td * 16 + (lane & 15);
  int rbase = ti * 16 + ((lane >> 4) << 3);
#pragma unroll
  for (int r = 0; r < 8; ++r)
    ctx[(rbase + r) * D + hd * HD + col] = (__bf16)acc[r];
}

// ---- K6: out = ctx @ Wout + bout (256x512, K=512) -> fp32 d_out -----------
__global__ void k_gemm_out(const __bf16* __restrict__ A,
                           const __bf16* __restrict__ Bt,
                           const float* __restrict__ bias,
                           float* __restrict__ C) {
  int wave = blockIdx.x * (blockDim.x >> 5) + (threadIdx.x >> 5);
  int lane = threadIdx.x & 31;
  int tm = wave / (D / 16), tn = wave % (D / 16);
  v8f acc = wmma_tile(A, D, Bt, D, tm * 16, tn * 16, D, lane);
  int col = tn * 16 + (lane & 15);
  float b = bias[col];
  int rbase = tm * 16 + ((lane >> 4) << 3);
#pragma unroll
  for (int r = 0; r < 8; ++r) C[(rbase + r) * D + col] = acc[r] + b;
}

// ---------------------------------------------------------------------------
extern "C" void kernel_launch(void* const* d_in, const int* in_sizes, int n_in,
                              void* d_out, int out_size, void* d_ws,
                              size_t ws_size, hipStream_t stream) {
  const float* x_hyp    = (const float*)d_in[0];
  const float* freqs    = (const float*)d_in[1];
  const float* c_sphere = (const float*)d_in[2];
  const float* Wqkv     = (const float*)d_in[3];
  const float* bqkv     = (const float*)d_in[4];
  const float* Wout     = (const float*)d_in[5];
  const float* bout     = (const float*)d_in[6];
  const float* c_logits = (const float*)d_in[7];
  const float* geo      = (const float*)d_in[8];
  float* out = (float*)d_out;

  char* ws = (char*)d_ws;
  size_t off = 0;
  auto alloc = [&](size_t bytes) {
    char* p = ws + off;
    off = (off + bytes + 255) & ~(size_t)255;
    return p;
  };
  __bf16* WqkvT = (__bf16*)alloc((size_t)QKVN * D * 2);
  __bf16* WoutT = (__bf16*)alloc((size_t)D * D * 2);
  __bf16* x_bf  = (__bf16*)alloc((size_t)N * D * 2);
  float*  qkv   = (float*) alloc((size_t)N * QKVN * 4);
  __bf16* q_hyp = (__bf16*)alloc((size_t)H * N * HD * 2);
  __bf16* k_hyp = (__bf16*)alloc((size_t)H * N * HD * 2);
  __bf16* vT    = (__bf16*)alloc((size_t)H * HD * N * 2);
  float*  q2    = (float*) alloc((size_t)H * N * 4);
  float*  k2    = (float*) alloc((size_t)H * N * 4);
  float*  Smat  = (float*) alloc((size_t)H * N * N * 4);
  __bf16* Wsm   = (__bf16*)alloc((size_t)H * N * N * 2);
  __bf16* ctx   = (__bf16*)alloc((size_t)N * D * 2);
  (void)ws_size; (void)in_sizes; (void)n_in; (void)out_size;

  // K0: weight prep (786432 elems)
  k_prep_weights<<<(QKVN * D + 255) / 256, 256, 0, stream>>>(Wqkv, Wout, WqkvT,
                                                             WoutT);
  // K1: logmap0
  k_logmap0<<<N, 256, 0, stream>>>(x_hyp, c_sphere, x_bf);
  // K2: QKV GEMM — 16*96 = 1536 waves
  k_gemm_qkv<<<1536 / 4, 128, 0, stream>>>(x_bf, WqkvT, bqkv, qkv);
  // K3: RoPE + expmap0
  k_rope_expmap<<<H * N, HD, 0, stream>>>(qkv, freqs, c_logits, q_hyp, k_hyp,
                                          vT, q2, k2);
  // K4a: Gram — 8 heads * 256 tiles = 2048 waves
  k_gram<<<2048 / 4, 128, 0, stream>>>(q_hyp, k_hyp, Smat);
  // K4b: distance + softmax
  k_scores_softmax<<<H * N, N, 0, stream>>>(Smat, q2, k2, c_logits, geo, Wsm);
  // K5: P@V — 8 heads * 64 tiles = 512 waves
  k_pv<<<512 / 4, 128, 0, stream>>>(Wsm, vT, ctx);
  // K6: output projection — 16*32 = 512 waves
  k_gemm_out<<<512 / 4, 128, 0, stream>>>(ctx, WoutT, bout, out);
}
